// FullModelTest_41601053229650
// MI455X (gfx1250) — 8-bit matrix operand order repaired and checked by reading, compiled, not run on hardware
//
#include <hip/hip_runtime.h>
#include <hip/hip_bf16.h>
#include <stdint.h>

#define NT      384
#define T_STEPS 4096
#define CTXF    6

typedef int   v8i __attribute__((ext_vector_type(8)));
typedef float v8f __attribute__((ext_vector_type(8)));

#define WMMA_FP8(a,b,c) __builtin_amdgcn_wmma_f32_16x16x64_fp8_fp8((a),(b),(short)0,(c),false,false)

#if defined(__HIP_DEVICE_COMPILE__) && __has_builtin(__builtin_amdgcn_global_load_async_to_lds_b128)
#define HAVE_ASYNC_LDS 1
typedef int v4ia __attribute__((ext_vector_type(4)));
typedef __attribute__((address_space(1))) v4ia* g128p;
typedef __attribute__((address_space(3))) v4ia* l128p;
#else
#define HAVE_ASYNC_LDS 0
#endif

enum {
  L1_WIH=0, L1_WHH, L1_BIH, L1_BHH,
  L2_WIH,   L2_WHH, L2_BIH, L2_BHH,
  L3_WIH,   L3_WHH, L3_BIH, L3_BHH,
  L4_WIH,   L4_WHH, L4_BIH, L4_BHH,
  F1_W1, F1_B1, F1_W2, F1_B2,
  F2_W1, F2_B1, F2_W2, F2_B2,
  F3_W1, F3_B1, F3_W2, F3_B2,
  H1_0, C1_0, H2_0, C2_0, H3_0, C3_0, H4_0, C4_0,
  TACT, ACT, CTXI, NIN
};

struct Params { const float* p[NIN]; float* out; };

#define L1_BYTES  (192*128)
#define L4_BYTES  (192*128)
#define F1B_BYTES (48*64)
#define F2A_BYTES (144*192)
#define F2B_BYTES (96*192)
#define F3A_BYTES (96*128)
#define F3B_BYTES (48*128)
#define OFF_L1   0
#define OFF_L4   (OFF_L1  + L1_BYTES)
#define OFF_F1B  (OFF_L4  + L4_BYTES)
#define OFF_F2A  (OFF_F1B + F1B_BYTES)
#define OFF_F2B  (OFF_F2A + F2A_BYTES)
#define OFF_F3A  (OFF_F2B + F2B_BYTES)
#define OFF_F3B  (OFF_F3A + F3A_BYTES)
#define LDSW_BYTES (OFF_F3B + F3B_BYTES)
#define OFF_L2F  LDSW_BYTES
#define OFF_L3F  (OFF_L2F + 24*3*1024)
#define WS_TOTAL (OFF_L3F + 36*5*1024)

struct __align__(1024) Smem {
  uint8_t w_l1 [L1_BYTES];
  uint8_t w_l4 [L4_BYTES];
  uint8_t w_f1b[F1B_BYTES];
  uint8_t w_f2a[F2A_BYTES];
  uint8_t w_f2b[F2B_BYTES];
  uint8_t w_f3a[F3A_BYTES];
  uint8_t w_f3b[F3B_BYTES];
  __align__(64) uint8_t x1v [128];
  __align__(64) uint8_t xf1 [64];
  __align__(64) uint8_t x2v [192];
  __align__(64) uint8_t x3v [320];
  __align__(64) uint8_t xh3 [192];
  __align__(64) uint8_t xh2 [192];
  __align__(64) uint8_t xf3a[128];
  __align__(64) uint8_t xf3b[128];
  __align__(64) uint8_t x4v [128];
  float b_l1[192], b_l2[384], b_l3[576], b_l4[192];
  float b_f1a[48], b_f1b[48], b_f2a[144], b_f2b[96], b_f3a[96], b_f3b[48];
  float w_f1a[144];
  float h1[48], c1[48], h2[96], c2[96], h3[144], c3[144], h4[48], c4[48];
  float out6[48];
  float gbuf[576];
  float fbuf[144];
  float tmp48[48];
  float f1hid[48];
  float avec[48];
};
static_assert(sizeof(Smem) < 160*1024, "LDS budget");

__device__ __forceinline__ unsigned char to_e4m3(float x){
  unsigned u = __float_as_uint(x);
  unsigned sg = (u >> 31) << 7;
  float a = fabsf(x);
  if (a != a)            return (unsigned char)(sg | 0x7f);
  if (a >= 448.f)        return (unsigned char)(sg | 0x7e);
  if (a < 0.0009765625f) return (unsigned char)sg;
  int e = (int)((__float_as_uint(a) >> 23) & 0xff) - 127;
  if (e < -6){
    int q = (int)rintf(a * 512.f);
    if (q >= 8) return (unsigned char)(sg | 0x08);
    return (unsigned char)(sg | (unsigned)q);
  }
  float sc = exp2f((float)(-e));
  int q = (int)rintf(a * sc * 8.f);
  if (q == 16){ q = 8; e++; }
  if (e > 8) return (unsigned char)(sg | 0x7e);
  return (unsigned char)(sg | ((unsigned)(e + 7) << 3) | (unsigned)(q - 8));
}

__device__ __forceinline__ float wval(const float* wih, const float* whh,
                                      int I, int H, int r, int k){
  if (k < I) return wih[r*I + k];
  int k2 = k - I;
  if (k2 < H) return whh[r*H + k2];
  return 0.f;
}

__device__ __forceinline__ int frag_kbase(int v, int g){
  return ((v >= 4) ? 32 : 0) + (v & 3) * 4 + g * 16;
}

__device__ __forceinline__ unsigned char pack_byte(const float* wih, const float* whh,
                                                   int I, int H, int KT, int p){
  int fi = p >> 10, rem = p & 1023;
  int ln = rem >> 5, j = rem & 31;
  int rt = fi / KT, kt = fi - rt*KT;
  int v = j >> 2, b = j & 3, g = ln >> 4;
  int k = kt*64 + frag_kbase(v, g) + b;
  int m = rt*16 + (ln & 15);
  return to_e4m3(wval(wih, whh, I, H, m, k));
}

__global__ void pack_weights(Params P, uint8_t* ws){
  const int stride = (int)(gridDim.x * blockDim.x);
  for (int p = (int)(blockIdx.x*blockDim.x + threadIdx.x); p < WS_TOTAL; p += stride){
    unsigned char v;
    if      (p < OFF_L4)  v = pack_byte(P.p[L1_WIH], P.p[L1_WHH], 48, 48, 2, p - OFF_L1);
    else if (p < OFF_F1B) v = pack_byte(P.p[L4_WIH], P.p[L4_WHH], 48, 48, 2, p - OFF_L4);
    else if (p < OFF_F2A) v = pack_byte(P.p[F1_W2],  nullptr,     48,  0, 1, p - OFF_F1B);
    else if (p < OFF_F2B) v = pack_byte(P.p[F2_W1],  nullptr,    144,  0, 3, p - OFF_F2A);
    else if (p < OFF_F3A) v = pack_byte(P.p[F2_W2],  nullptr,    144,  0, 3, p - OFF_F2B);
    else if (p < OFF_F3B) v = pack_byte(P.p[F3_W1],  nullptr,     96,  0, 2, p - OFF_F3A);
    else if (p < OFF_L2F) v = pack_byte(P.p[F3_W2],  nullptr,     96,  0, 2, p - OFF_F3B);
    else if (p < OFF_L3F) v = pack_byte(P.p[L2_WIH], P.p[L2_WHH], 96, 96, 3, p - OFF_L2F);
    else                  v = pack_byte(P.p[L3_WIH], P.p[L3_WHH],144,144, 5, p - OFF_L3F);
    ws[p] = v;
  }
}

__device__ __forceinline__ v8i load_bvec(const uint8_t* xs, int kt, int lane){
  const int g = (lane >> 4) & 1;
  const int4* p = (const int4*)(xs + kt*64 + g*16);
  int4 lo = p[0], hi = p[2];
  v8i b;
  b[0]=lo.x; b[1]=lo.y; b[2]=lo.z; b[3]=lo.w;
  b[4]=hi.x; b[5]=hi.y; b[6]=hi.z; b[7]=hi.w;
  return b;
}

__device__ __forceinline__ v8f bias_c(const float* b, int rt, int lane){
  const float* p = b + rt*16 + ((lane >> 4) & 1) * 8;
  v8f c;
  #pragma unroll
  for (int r = 0; r < 8; r++) c[r] = p[r];
  return c;
}

__device__ __forceinline__ void store_d(float* out, int rt, int lane, v8f c){
  if ((lane & 15) == 0){
    float* p = out + rt*16 + ((lane >> 4) & 1) * 8;
    #pragma unroll
    for (int r = 0; r < 8; r++) p[r] = c[r];
  }
}

template<int KT>
__device__ __forceinline__ void matvec_lds(const uint8_t* w, const uint8_t* xs,
                                           const float* bias, float* out,
                                           int nrt, int wv, int lane){
  if (wv < nrt){
    v8f c = bias_c(bias, wv, lane);
    #pragma unroll
    for (int kt = 0; kt < KT; kt++){
      v8i a = *(const v8i*)(w + (wv*KT + kt)*1024 + lane*32);
      v8i b = load_bvec(xs, kt, lane);
      c = WMMA_FP8(a, b, c);
    }
    store_d(out, wv, lane, c);
  }
}

__device__ __forceinline__ float sigmf(float x){ return 1.f / (1.f + __expf(-x)); }

__device__ __forceinline__ void lstm_nl(const float* g, float* h, float* c, int H, int tid){
  if (tid < H){
    float i = g[tid], f = g[H+tid], gg = g[2*H+tid], o = g[3*H+tid];
    float cn = sigmf(f)*c[tid] + sigmf(i)*tanhf(gg);
    float hn = sigmf(o)*tanhf(cn);
    c[tid] = cn; h[tid] = hn;
  }
}

__global__ __launch_bounds__(NT, 1)
void rnn_persistent(Params P, const uint8_t* ws){
  __shared__ Smem s;
  const int tid  = threadIdx.x;
  const int wv   = tid >> 5;
  const int lane = tid & 31;

  {
    uint8_t* dst = s.w_l1;
#if HAVE_ASYNC_LDS
    #pragma unroll
    for (int i = 0; i < LDSW_BYTES/16/NT; i++){
      int idx = (i*NT + tid) * 16;
      __builtin_amdgcn_global_load_async_to_lds_b128((g128p)(ws + idx),
                                                     (l128p)(dst + idx), 0, 0);
    }
#if __has_builtin(__builtin_amdgcn_s_wait_asynccnt)
    __builtin_amdgcn_s_wait_asynccnt(0);
#else
    asm volatile("s_wait_asynccnt 0" ::: "memory");
#endif
#else
    for (int i = tid; i < LDSW_BYTES/16; i += NT)
      ((int4*)dst)[i] = ((const int4*)ws)[i];
#endif
  }

  for (int j = tid; j < 192; j += NT) s.b_l1[j] = P.p[L1_BIH][j] + P.p[L1_BHH][j];
  for (int j = tid; j < 384; j += NT) s.b_l2[j] = P.p[L2_BIH][j] + P.p[L2_BHH][j];
  for (int j = tid; j < 576; j += NT) s.b_l3[j] = P.p[L3_BIH][j] + P.p[L3_BHH][j];
  for (int j = tid; j < 192; j += NT) s.b_l4[j] = P.p[L4_BIH][j] + P.p[L4_BHH][j];
  for (int j = tid; j < 48;  j += NT){ s.b_f1a[j]=P.p[F1_B1][j]; s.b_f1b[j]=P.p[F1_B2][j]; s.b_f3b[j]=P.p[F3_B2][j]; }
  for (int j = tid; j < 144; j += NT){ s.b_f2a[j]=P.p[F2_B1][j]; s.w_f1a[j]=P.p[F1_W1][j]; }
  for (int j = tid; j < 96;  j += NT){ s.b_f2b[j]=P.p[F2_B2][j]; s.b_f3a[j]=P.p[F3_B1][j]; }
  for (int j = tid; j < 48;  j += NT){ s.h1[j]=P.p[H1_0][j]; s.c1[j]=P.p[C1_0][j];
                                       s.h4[j]=P.p[H4_0][j]; s.c4[j]=P.p[C4_0][j]; s.out6[j]=0.f; }
  for (int j = tid; j < 96;  j += NT){ s.h2[j]=P.p[H2_0][j]; s.c2[j]=P.p[C2_0][j]; }
  for (int j = tid; j < 144; j += NT){ s.h3[j]=P.p[H3_0][j]; s.c3[j]=P.p[C3_0][j]; }

  v8i w2f[2][3], w3f[3][5];
  {
    const uint8_t* l2f = ws + OFF_L2F;
    const uint8_t* l3f = ws + OFF_L3F;
    #pragma unroll
    for (int r = 0; r < 2; r++)
      #pragma unroll
      for (int kt = 0; kt < 3; kt++)
        w2f[r][kt] = *(const v8i*)(l2f + ((wv*2 + r)*3 + kt)*1024 + lane*32);
    #pragma unroll
    for (int r = 0; r < 3; r++)
      #pragma unroll
      for (int kt = 0; kt < 5; kt++)
        w3f[r][kt] = *(const v8i*)(l3f + ((wv*3 + r)*5 + kt)*1024 + lane*32);
  }

  __syncthreads();

  const float* tact = P.p[TACT];
  const float* actp = P.p[ACT];
  const float* ctxp = P.p[CTXI];

  for (int t = 0; t < T_STEPS; t++){
    if (tid < 48){
      float a0 = actp[t*3+0], a1 = actp[t*3+1], a2 = actp[t*3+2];
      float v = s.w_f1a[tid*3]*a0 + s.w_f1a[tid*3+1]*a1 + s.w_f1a[tid*3+2]*a2 + s.b_f1a[tid];
      s.f1hid[tid] = fmaxf(v, 0.f);
    }
    const bool fb = (t > CTXF - 1);
    for (int k = tid; k < 128; k += NT){
      float v = 0.f;
      if (k < 48)      v = fb ? s.out6[k] : tact[t*48 + k];
      else if (k < 96) v = s.h1[k - 48];
      s.x1v[k] = to_e4m3(v);
    }
    __syncthreads();
    for (int k = tid; k < 64; k += NT) s.xf1[k] = (k < 48) ? to_e4m3(s.f1hid[k]) : 0;
    __syncthreads();
    matvec_lds<2>(s.w_l1,  s.x1v, s.b_l1,  s.gbuf,  12, wv, lane);
    matvec_lds<1>(s.w_f1b, s.xf1, s.b_f1b, s.tmp48,  3, wv, lane);
    __syncthreads();
    lstm_nl(s.gbuf, s.h1, s.c1, 48, tid);
    if (tid >= 64 && tid < 112) s.avec[tid-64] = fmaxf(s.tmp48[tid-64], 0.f);
    if (t + 1 < T_STEPS){
      if (tid == 128) __builtin_prefetch(actp + (t+1)*3, 0, 3);
      if (tid == 129) __builtin_prefetch(ctxp + (t+1)*48, 0, 3);
      if (tid == 130) __builtin_prefetch(ctxp + (t+1)*48 + 32, 0, 3);
      if (tid == 131 && (t+1) < CTXF) __builtin_prefetch(tact + (t+1)*48, 0, 3);
    }
    __syncthreads();
    for (int k = tid; k < 192; k += NT){
      float v;
      if (k < 48)       v = ctxp[t*48 + k];
      else if (k < 96)  v = s.h1[k - 48];
      else              v = s.h2[k - 96];
      s.x2v[k] = to_e4m3(v);
    }
    __syncthreads();
    {
      v8f c0 = bias_c(s.b_l2, wv*2 + 0, lane);
      v8f c1 = bias_c(s.b_l2, wv*2 + 1, lane);
      #pragma unroll
      for (int kt = 0; kt < 3; kt++){
        v8i b = load_bvec(s.x2v, kt, lane);
        c0 = WMMA_FP8(w2f[0][kt], b, c0);
        c1 = WMMA_FP8(w2f[1][kt], b, c1);
      }
      store_d(s.gbuf, wv*2 + 0, lane, c0);
      store_d(s.gbuf, wv*2 + 1, lane, c1);
    }
    __syncthreads();
    lstm_nl(s.gbuf, s.h2, s.c2, 96, tid);
    __syncthreads();
    for (int k = tid; k < 320; k += NT){
      float v = 0.f;
      if (k < 48)       v = s.avec[k];
      else if (k < 144) v = s.h2[k - 48];
      else if (k < 288) v = s.h3[k - 144];
      s.x3v[k] = to_e4m3(v);
    }
    __syncthreads();
    {
      v8f c0 = bias_c(s.b_l3, wv*3 + 0, lane);
      v8f c1 = bias_c(s.b_l3, wv*3 + 1, lane);
      v8f c2 = bias_c(s.b_l3, wv*3 + 2, lane);
      #pragma unroll
      for (int kt = 0; kt < 5; kt++){
        v8i b = load_bvec(s.x3v, kt, lane);
        c0 = WMMA_FP8(w3f[0][kt], b, c0);
        c1 = WMMA_FP8(w3f[1][kt], b, c1);
        c2 = WMMA_FP8(w3f[2][kt], b, c2);
      }
      store_d(s.gbuf, wv*3 + 0, lane, c0);
      store_d(s.gbuf, wv*3 + 1, lane, c1);
      store_d(s.gbuf, wv*3 + 2, lane, c2);
    }
    __syncthreads();
    lstm_nl(s.gbuf, s.h3, s.c3, 144, tid);
    __syncthreads();
    for (int k = tid; k < 192; k += NT) s.xh3[k] = (k < 144) ? to_e4m3(s.h3[k]) : 0;
    __syncthreads();
    matvec_lds<3>(s.w_f2a, s.xh3, s.b_f2a, s.fbuf, 9, wv, lane);
    __syncthreads();
    for (int k = tid; k < 192; k += NT) s.xh2[k] = (k < 144) ? to_e4m3(fmaxf(s.fbuf[k], 0.f)) : 0;
    __syncthreads();
    matvec_lds<3>(s.w_f2b, s.xh2, s.b_f2b, s.fbuf, 6, wv, lane);
    __syncthreads();
    for (int k = tid; k < 128; k += NT) s.xf3a[k] = (k < 96) ? to_e4m3(fmaxf(s.fbuf[k], 0.f)) : 0;
    __syncthreads();
    matvec_lds<2>(s.w_f3a, s.xf3a, s.b_f3a, s.fbuf, 6, wv, lane);
    __syncthreads();
    for (int k = tid; k < 128; k += NT) s.xf3b[k] = (k < 96) ? to_e4m3(fmaxf(s.fbuf[k], 0.f)) : 0;
    __syncthreads();
    matvec_lds<2>(s.w_f3b, s.xf3b, s.b_f3b, s.fbuf, 3, wv, lane);
    __syncthreads();
    for (int k = tid; k < 128; k += NT){
      float v = 0.f;
      if (k < 48)      v = fmaxf(s.fbuf[k], 0.f);
      else if (k < 96) v = s.h4[k - 48];
      s.x4v[k] = to_e4m3(v);
    }
    __syncthreads();
    matvec_lds<2>(s.w_l4, s.x4v, s.b_l4, s.gbuf, 12, wv, lane);
    __syncthreads();
    if (tid < 48){
      float i = s.gbuf[tid], f = s.gbuf[48+tid], gg = s.gbuf[96+tid], o = s.gbuf[144+tid];
      float cn = sigmf(f)*s.c4[tid] + sigmf(i)*tanhf(gg);
      float hn = sigmf(o)*tanhf(cn);
      s.c4[tid] = cn; s.h4[tid] = hn; s.out6[tid] = hn;
      P.out[t*48 + tid] = hn;
    }
    __syncthreads();
  }
}

extern "C" void kernel_launch(void* const* d_in, const int* in_sizes, int n_in,
                              void* d_out, int out_size, void* d_ws, size_t ws_size,
                              hipStream_t stream) {
  (void)in_sizes; (void)n_in; (void)out_size; (void)ws_size;
  Params P;
  for (int i = 0; i < NIN; i++) P.p[i] = (const float*)d_in[i];
  P.out = (float*)d_out;
  uint8_t* ws = (uint8_t*)d_ws;
  pack_weights<<<dim3(256), dim3(256), 0, stream>>>(P, ws);
  rnn_persistent<<<dim3(1), dim3(NT), 0, stream>>>(P, ws);
}
